// NeighborhoodCrossAttention_54228257079965
// MI455X (gfx1250) — compile-verified, matrix operand order checked by reading, not run on hardware
//
#include <hip/hip_runtime.h>
#include <hip/hip_bf16.h>
#include <math.h>

typedef __attribute__((ext_vector_type(16))) _Float16 v16h;
typedef __attribute__((ext_vector_type(8)))  _Float16 v8h;
typedef __attribute__((ext_vector_type(8)))  float    v8f;

#define NW    11
#define NN    121
#define S_TOT 484      // HEADS * N * N
#define K_PAD 512      // K padded to multiple of 32
#define DIM   128
#define TP    16       // pixels per workgroup (WMMA M)
#define HH    64
#define WW    64

// ---- Prep: transpose mlp_w [484 x 128] f32 -> Wt [128 x 512] f16, zero-padded K ----
__global__ __launch_bounds__(256)
void prep_w_kernel(const float* __restrict__ mlp_w, _Float16* __restrict__ Wt)
{
    const int idx = blockIdx.x * 256 + threadIdx.x;   // 128*512 = 65536 elems
    const int col = idx >> 9;                         // 0..127
    const int k   = idx & 511;                        // 0..511
    Wt[col * K_PAD + k] = (k < S_TOT) ? (_Float16)mlp_w[k * DIM + col] : (_Float16)0.f;
}

__global__ __launch_bounds__(256)
void nca_fused_kernel(const float* __restrict__ q,
                      const float* __restrict__ kv,
                      const float* __restrict__ pos,
                      const _Float16* __restrict__ Wt,   // [128][512] f16, padded
                      const float* __restrict__ mlp_b,
                      const float* __restrict__ ln_g,
                      const float* __restrict__ ln_b,
                      float* __restrict__ out)
{
    __shared__ __align__(16) float    q_lds[TP][DIM];     // 8 KB
    __shared__ __align__(16) _Float16 A_lds[TP][K_PAD];   // 16 KB, scores (f16)
    __shared__ __align__(16) float    X_lds[TP][DIM];     // 8 KB, pre-LN activations
    __shared__ float mu_lds[TP];
    __shared__ float rs_lds[TP];

    const int tid  = threadIdx.x;
    const int lane = tid & 31;          // wave32
    const int wv   = tid >> 5;          // 8 waves
    const int half = lane >> 4;

    // 512 workgroups: 16-pixel strip along W
    const int wg = blockIdx.x;
    const int b  = wg >> 8;             // 256 tiles per batch image
    const int r  = wg & 255;
    const int x  = r >> 2;
    const int y0 = (r & 3) << 4;
    const int pbase = (b * HH + x) * WW + y0;   // global pixel index of p=0

    // ---- Phase 0: cooperative q load + zero A-tile K padding ----
    for (int t = tid; t < TP * DIM; t += 256)
        q_lds[t >> 7][t & 127] = q[(pbase + (t >> 7)) * DIM + (t & 127)];
    for (int t = tid; t < TP * (K_PAD - S_TOT); t += 256)
        A_lds[t / (K_PAD - S_TOT)][S_TOT + t % (K_PAD - S_TOT)] = (_Float16)0.f;
    __syncthreads();

    // ---- Phase A: scores  s[p, h*121+n] -> A_lds (f16) ----
    // flat = s*32 + d  ->  cell m = s>>2, chunk c0 = (s&3)*32, head h = s/121
    for (int idx = tid; idx < TP * S_TOT; idx += 256) {
        const int p  = idx & 15;
        const int s  = idx >> 4;
        const int m  = s >> 2;
        const int c0 = (s & 3) << 5;
        const int h  = s / NN;
        const int i  = m / NW;
        const int j  = m - i * NW;
        const int xx = x + i - 5;
        const int yy = y0 + p + j - 5;
        const bool inb = (xx >= 0) & (xx < HH) & (yy >= 0) & (yy < WW);

        const float4* pv4 = (const float4*)(pos + (i * NW + j) * DIM + c0);
        const float4* qv4 = (const float4*)(&q_lds[p][h << 5]);
        float acc = 0.f;
        if (inb) {
            const float4* kv4 = (const float4*)(kv + ((b * HH + xx) * WW + yy) * DIM + c0);
            #pragma unroll
            for (int d = 0; d < 8; ++d) {
                const float4 a = kv4[d], pp = pv4[d], qq = qv4[d];
                acc += (a.x + pp.x) * qq.x + (a.y + pp.y) * qq.y
                     + (a.z + pp.z) * qq.z + (a.w + pp.w) * qq.w;
            }
        } else {                        // padded kv == 0, pos_enc still added
            #pragma unroll
            for (int d = 0; d < 8; ++d) {
                const float4 pp = pv4[d], qq = qv4[d];
                acc += pp.x * qq.x + pp.y * qq.y + pp.z * qq.z + pp.w * qq.w;
            }
        }
        A_lds[p][s] = (_Float16)acc;
    }
    __syncthreads();

    // ---- Phase B: [16 x 512] @ [512 x 128] via v_wmma_f32_16x16x32_f16 ----
    const int n0  = wv << 4;            // each wave owns 16 output channels
    const int col = n0 + (lane & 15);
    const int row = lane & 15;
    const _Float16* Wcol = Wt + col * K_PAD;   // lane's column, K-contiguous

    v8f c = {};
    #pragma unroll
    for (int kk = 0; kk < K_PAD / 32; ++kk) {
        // A fragment: elems 0..7 = K half*8..+7, elems 8..15 = K 16+half*8..+7
        const _Float16* Ab = &A_lds[row][kk * 32 + half * 8];
        const v8h alo = *(const v8h*)Ab;               // ds_load_b128
        const v8h ahi = *(const v8h*)(Ab + 16);        // ds_load_b128
        const v16h a = __builtin_shufflevector(alo, ahi,
                         0,1,2,3,4,5,6,7,8,9,10,11,12,13,14,15);

        // B fragment: same K-run layout, unconditional global_load_b128 x2
        const v8h blo = *(const v8h*)(Wcol + kk * 32 + half * 8);
        const v8h bhi = *(const v8h*)(Wcol + kk * 32 + 16 + half * 8);
        const v16h bb = __builtin_shufflevector(blo, bhi,
                         0,1,2,3,4,5,6,7,8,9,10,11,12,13,14,15);

        c = __builtin_amdgcn_wmma_f32_16x16x32_f16(false, a, false, bb,
                                                   (short)0, c, false, false);
    }

    // ---- Epilogue: bias + exact GELU + residual, C layout M=v+8*half, N=lane&15 ----
    {
        const float bias = mlp_b[col];
        #pragma unroll
        for (int v = 0; v < 8; ++v) {
            const int p = v + (half << 3);
            float val = c[v] + bias;
            val = 0.5f * val * (1.f + erff(val * 0.70710678118654752f));
            val += q_lds[p][col];
            X_lds[p][col] = val;
        }
    }
    __syncthreads();

    // ---- Phase C: LayerNorm over 128 channels ----
    if (tid < TP) {
        float sum = 0.f;
        #pragma unroll 4
        for (int cc = 0; cc < DIM; ++cc) sum += X_lds[tid][cc];
        const float mu = sum * (1.f / DIM);
        float var = 0.f;
        #pragma unroll 4
        for (int cc = 0; cc < DIM; ++cc) {
            const float d = X_lds[tid][cc] - mu;
            var += d * d;
        }
        mu_lds[tid] = mu;
        rs_lds[tid] = rsqrtf(var * (1.f / DIM) + 1e-5f);
    }
    __syncthreads();

    for (int t = tid; t < TP * DIM; t += 256) {
        const int p = t >> 7, cc = t & 127;
        out[(pbase + p) * DIM + cc] =
            (X_lds[p][cc] - mu_lds[p]) * rs_lds[p] * ln_g[cc] + ln_b[cc];
    }
}

extern "C" void kernel_launch(void* const* d_in, const int* in_sizes, int n_in,
                              void* d_out, int out_size, void* d_ws, size_t ws_size,
                              hipStream_t stream) {
    const float* q     = (const float*)d_in[0];
    const float* kv    = (const float*)d_in[1];
    const float* pos   = (const float*)d_in[2];
    const float* mlp_w = (const float*)d_in[3];
    const float* mlp_b = (const float*)d_in[4];
    const float* ln_g  = (const float*)d_in[5];
    const float* ln_b  = (const float*)d_in[6];
    float* out = (float*)d_out;
    _Float16* Wt = (_Float16*)d_ws;     // 128*512*2 = 128 KB scratch

    hipLaunchKernelGGL(prep_w_kernel, dim3(256), dim3(256), 0, stream, mlp_w, Wt);
    hipLaunchKernelGGL(nca_fused_kernel, dim3(512), dim3(256), 0, stream,
                       q, kv, pos, Wt, mlp_b, ln_g, ln_b, out);
}
